// UseRNN_9534827397492
// MI455X (gfx1250) — matrix operand order repaired and checked by reading, not compiled, not run on hardware
//
#include <hip/hip_runtime.h>
#include <hip/hip_bf16.h>
#include <math.h>
#include <stdint.h>

// Problem constants (match reference)
#define TT 512
#define BS 64
#define II 256
#define HH 512
#define OO 256
#define TBDIM (TT * BS)   // 32768 rows for input projection

typedef __bf16 bf16_t;
typedef __attribute__((ext_vector_type(16))) __bf16 v16bf;
typedef __attribute__((ext_vector_type(8)))  float  v8f;

// Native CDNA5 V_TANH_F32 when the toolchain exposes it (serial critical path!)
#if __has_builtin(__builtin_amdgcn_tanhf)
#define FAST_TANH(x) __builtin_amdgcn_tanhf(x)
#elif __has_builtin(__builtin_amdgcn_tanh_f32)
#define FAST_TANH(x) __builtin_amdgcn_tanh_f32(x)
#else
#define FAST_TANH(x) tanhf(x)
#endif

// CDNA5 async LDS->global store path (ASYNCcnt-tracked, no VGPR round trip)
#if __has_builtin(__builtin_amdgcn_global_store_async_from_lds_b128)
#define HAVE_ASYNC_LDS 1
typedef int v4i_t __attribute__((vector_size(16)));
typedef __attribute__((address_space(3))) v4i_t as3_v4i;
#else
#define HAVE_ASYNC_LDS 0
#endif

#if __has_builtin(__builtin_amdgcn_s_wait_asynccnt)
#define WAIT_ASYNC(n) __builtin_amdgcn_s_wait_asynccnt(n)
#else
#define WAIT_ASYNC(n) asm volatile("s_wait_asynccnt %0" ::"i"(n))
#endif

// ---------------------------------------------------------------------------
// Fragment index helpers (CDNA5 16-bit WMMA VGPR layouts, CDNA5 ISA)
// A (16x32, MxK): lane l<16 row M=l holds K {0..7,16..23}; lane l+16 holds
//                 K {8..15,24..31}.  K(e,hi) = (e&7) + ((e>>3)<<4) + hi*8
// B (32x16, KxN): lane l<16 col N=l holds K {0..7,16..23}; lane l+16 holds
//                 K {8..15,24..31}.  K(e,hi) = (e&7) + ((e>>3)<<4) + hi*8
// C/D (16x16 f32): lane l<16: vgpr r -> (M=r,   N=l)
//                  lane l+16: vgpr r -> (M=r+8, N=l-16)
// ---------------------------------------------------------------------------
__device__ __forceinline__ int ka_idx(int e, int hi) { return (e & 7) + ((e >> 3) << 4) + hi * 8; }
__device__ __forceinline__ int kb_idx(int e, int hi) { return (e & 7) + ((e >> 3) << 4) + hi * 8; }

__device__ __forceinline__ v8f wmma_bf16(v16bf a, v16bf b, v8f c) {
  return __builtin_amdgcn_wmma_f32_16x16x32_bf16(false, a, false, b, (short)0, c, false, false);
}

// ---------------------------------------------------------------------------
// Pack x [TB, I] f32 -> A-fragment-swizzled bf16 [TB/16][I/32][lane=32][16]
// One thread per (mt, kt, lane); writes 32 contiguous bytes.
// ---------------------------------------------------------------------------
__global__ void pack_x_kernel(const float* __restrict__ x, bf16_t* __restrict__ xsw) {
  int g    = blockIdx.x * blockDim.x + threadIdx.x;     // (mt*8 + kt)*32 + lane
  int lane = g & 31;
  int kt   = (g >> 5) & 7;                              // I/32 = 8
  int mt   = g >> 8;
  if (mt >= TBDIM / 16) return;
  int l = lane & 15, hi = lane >> 4;
  int row = mt * 16 + l;
  v16bf frag;
#pragma unroll
  for (int e = 0; e < 16; ++e) {
    frag[e] = (bf16_t)x[(size_t)row * II + kt * 32 + ka_idx(e, hi)];
  }
  *(v16bf*)(xsw + (size_t)g * 16) = frag;
}

// ---------------------------------------------------------------------------
// Pack weight W [N, K] f32 -> B-fragment-swizzled bf16 [N/16][K/32][lane=32][16]
// out[b][n] = sum_k in[b][k] * W[n][k]   (B element (k,n) = W[n][k])
// ---------------------------------------------------------------------------
__global__ void pack_w_kernel(const float* __restrict__ W, bf16_t* __restrict__ dst,
                              int N, int K) {
  int g    = blockIdx.x * blockDim.x + threadIdx.x;
  int lane = g & 31;
  int rest = g >> 5;
  int kt   = rest % (K / 32);
  int nt   = rest / (K / 32);
  if (nt >= N / 16) return;
  int l = lane & 15, hi = lane >> 4;
  int nrow = nt * 16 + l;
  v16bf frag;
#pragma unroll
  for (int e = 0; e < 16; ++e) {
    frag[e] = (bf16_t)W[(size_t)nrow * K + kt * 32 + kb_idx(e, hi)];
  }
  *(v16bf*)(dst + (size_t)g * 16) = frag;
}

// ---------------------------------------------------------------------------
// Input projection: xp = x . W_ih^T + b_ih + b_hh, stored in C-FRAGMENT order
// xp_swz[mt_g(2048)][nt(32)][lane(32)][r(8)]  (f32, 32B contiguous per thread)
// so the recurrence epilogue reads its accumulator image with one vector load.
// 8 waves / block, one 16x16 tile per wave, K-loop over I in steps of 32.
// ---------------------------------------------------------------------------
__global__ void input_proj_kernel(const bf16_t* __restrict__ xsw,
                                  const bf16_t* __restrict__ wih,
                                  const float* __restrict__ b_ih,
                                  const float* __restrict__ b_hh,
                                  float* __restrict__ xp) {
  int wave = threadIdx.x >> 5, lane = threadIdx.x & 31;
  int l = lane & 15;
  int tile = blockIdx.x * 8 + wave;                     // 2048*32 tiles
  int mt = tile >> 5;                                   // TB tile
  int nt = tile & 31;                                   // H tile
  const v16bf* A  = (const v16bf*)xsw + (size_t)(mt * 8) * 32 + lane;
  const v16bf* Bp = (const v16bf*)wih + (size_t)(nt * 8) * 32 + lane;
  v8f acc = {};
#pragma unroll
  for (int kt = 0; kt < 8; ++kt) {
    v16bf a = A[kt * 32];
    v16bf b = Bp[kt * 32];
    acc = wmma_bf16(a, b, acc);
  }
  int col = nt * 16 + l;                                // same col for all 8 rows
  acc = acc + (b_ih[col] + b_hh[col]);                  // vector splat add
  *(v8f*)(xp + (((size_t)mt * 32 + nt) * 32 + lane) * 8) = acc;
}

// ---------------------------------------------------------------------------
// Recurrence: single workgroup (32 waves), h double-buffered in LDS in
// A-fragment-swizzled layout (64KB each). Per step: h_new = tanh(xp_t + h Whh^T)
// Wave w: m-tile = w&3 (B dim), 4 n-tiles = (w>>2)*4.. (H dim). K = 512.
// xp is read in C-fragment order (one 32B vector load per tile); h_new is
// scattered only into LDS, then the wave's own contiguous 2KB slice of hn is
// bulk-copied to global rnn staging via async LDS->global stores (ASYNCcnt),
// overlapping the next step's WMMA work.  t == T-1 also writes f32 h_last.
// ---------------------------------------------------------------------------
__global__ void rnn_recurrence_kernel(const float* __restrict__ xp,
                                      const bf16_t* __restrict__ whh,
                                      bf16_t* __restrict__ rnn_swz,
                                      float* __restrict__ h_out) {
  extern __shared__ v16bf smemv[];                       // 2 * 64KB, 32B aligned
  bf16_t* hbuf = (bf16_t*)smemv;                         // [2][4][16][32][16]
  const int HS = 4 * 16 * 32 * 16;                       // 32768 bf16 per buffer

  int tid = threadIdx.x;
  int wave = tid >> 5, lane = tid & 31;
  int l = lane & 15, hi = lane >> 4;
  int mt = wave & 3;
  int ng = wave >> 2;                                    // 0..7 -> n-tiles ng*4..ng*4+3
  // this wave's epilogue region inside a h buffer (exactly the bytes it writes)
  const int chunk = mt * 8192 + ng * 1024 + lane * 32;   // bf16 elements, 64B/lane

  // h0 = 0
  for (int i = tid; i < HS; i += 1024) hbuf[i] = (bf16_t)0.0f;
  __syncthreads();

  for (int t = 0; t < TT; ++t) {
    const bf16_t* hc = hbuf + (t & 1) * HS;
    bf16_t*       hn = hbuf + ((t & 1) ^ 1) * HS;
    // xp tile base for this step, C-fragment layout [t*4+mt][nt][lane][8]
    const float* xpt = xp + (((size_t)(t * 4 + mt) * 32) * 32 + lane) * 8;
    __builtin_prefetch(xpt, 0, 0);                       // global_prefetch_b8

    v8f acc[4] = {};
    const v16bf* A = (const v16bf*)hc + (size_t)(mt * 16) * 32 + lane;
#pragma unroll 4
    for (int kt = 0; kt < 16; ++kt) {
      v16bf a = A[kt * 32];                              // ds_load_b128 x2
#pragma unroll
      for (int j = 0; j < 4; ++j) {
        int nt = ng * 4 + j;
        v16bf b = *((const v16bf*)whh + (size_t)(nt * 16 + kt) * 32 + lane);
        acc[j] = wmma_bf16(a, b, acc[j]);
      }
    }

#if HAVE_ASYNC_LDS
    // hn buffer was async-stored 2 steps ago (4 b128 per thread, in-order):
    // leaving <=4 outstanding guarantees that batch has completed.
    WAIT_ASYNC(4);
#endif

#pragma unroll
    for (int j = 0; j < 4; ++j) {
      int nt  = ng * 4 + j;
      int col = nt * 16 + l;
      v8f xv = *(const v8f*)(xpt + (size_t)nt * 32 * 8); // global_load_b128 x2
#pragma unroll
      for (int r = 0; r < 8; ++r) {
        int row = mt * 16 + hi * 8 + r;
        float v = FAST_TANH(acc[j][r] + xv[r]);          // v_tanh_f32
        // A-swizzled position of (row, col) for next step / output proj
        int kt2   = col >> 5;
        int kk    = col & 31;
        int hi2   = (kk >> 3) & 1;
        int e2    = (kk & 7) + ((kk >> 4) << 3);
        int lane2 = (row & 15) + (hi2 << 4);
        int idx   = ((mt * 16 + kt2) * 32 + lane2) * 16 + e2;
        hn[idx] = (bf16_t)v;                             // ds_store_b16 (scatter)
        if (t == TT - 1) h_out[(size_t)row * HH + col] = v;
      }
    }
    __syncthreads();

    // Bulk-copy this wave's own 2KB slice of hn -> rnn_swz[t] (A-layout image)
    {
      bf16_t* lsrc = hn + chunk;
      bf16_t* gdst = rnn_swz + (size_t)t * HS + chunk;
#if HAVE_ASYNC_LDS
#pragma unroll
      for (int k2 = 0; k2 < 4; ++k2) {                   // 4 x 16B per thread
        __builtin_amdgcn_global_store_async_from_lds_b128(
            (v4i_t*)(gdst + k2 * 8),
            (as3_v4i*)(uint32_t)(uintptr_t)(lsrc + k2 * 8), 0, 0);
      }
#else
      *(v16bf*)gdst         = *(const v16bf*)lsrc;       // 2 x 32B fallback copy
      *((v16bf*)gdst + 1)   = *((const v16bf*)lsrc + 1);
#endif
    }
  }
#if HAVE_ASYNC_LDS
  WAIT_ASYNC(0);
#endif
}

// ---------------------------------------------------------------------------
// Output projection: y[tb,o] = rnn[tb,:] . W_out[o,:] + b_out[o]
// 2 blocks per timestep (O split in halves), 8 waves, 4 tiles/wave, K = 512.
// ---------------------------------------------------------------------------
__global__ void output_proj_kernel(const bf16_t* __restrict__ rnn_swz,
                                   const bf16_t* __restrict__ wout,
                                   const float* __restrict__ b_out,
                                   float* __restrict__ y) {
  int wave = threadIdx.x >> 5, lane = threadIdx.x & 31;
  int l = lane & 15, hi = lane >> 4;
  int t    = blockIdx.x >> 1;
  int half = blockIdx.x & 1;
  int mt = wave & 3;
  int ng = wave >> 2;                                    // 0..1

  const v16bf* A  = (const v16bf*)rnn_swz + (((size_t)t * 4 + mt) * 16) * 32 + lane;
  const v16bf* Bp = (const v16bf*)wout + lane;
  v8f acc[4] = {};
#pragma unroll 4
  for (int kt = 0; kt < 16; ++kt) {
    v16bf a = A[kt * 32];
#pragma unroll
    for (int j = 0; j < 4; ++j) {
      int nt = half * 8 + ng * 4 + j;
      v16bf b = Bp[(size_t)(nt * 16 + kt) * 32];
      acc[j] = wmma_bf16(a, b, acc[j]);
    }
  }
#pragma unroll
  for (int j = 0; j < 4; ++j) {
    int nt  = half * 8 + ng * 4 + j;
    int col = nt * 16 + l;
#pragma unroll
    for (int r = 0; r < 8; ++r) {
      int row = mt * 16 + hi * 8 + r;
      y[((size_t)t * BS + row) * OO + col] = acc[j][r] + b_out[col];
    }
  }
}

// ---------------------------------------------------------------------------
// Workspace layout (bytes):
//   xsw     @ 0         : 16 MB  (bf16, 2048*8*512)
//   wih_sw  @ 16777216  : 256 KB
//   whh_sw  @ 17039360  : 512 KB
//   wout_sw @ 17563648  : 256 KB
//   xp      @ 17825792  : 64 MB  (f32, C-fragment swizzled)
//   rnn_swz @ 84934656  : 32 MB  (bf16)   total ~113 MB
// ---------------------------------------------------------------------------
extern "C" void kernel_launch(void* const* d_in, const int* in_sizes, int n_in,
                              void* d_out, int out_size, void* d_ws, size_t ws_size,
                              hipStream_t stream) {
  const float* x     = (const float*)d_in[0];
  const float* W_ih  = (const float*)d_in[1];
  const float* W_hh  = (const float*)d_in[2];
  const float* b_ih  = (const float*)d_in[3];
  const float* b_hh  = (const float*)d_in[4];
  const float* W_out = (const float*)d_in[5];
  const float* b_out = (const float*)d_in[6];

  float* y     = (float*)d_out;
  float* h_out = y + (size_t)TT * BS * OO;               // h_last tail [B, H]

  char* ws = (char*)d_ws;
  bf16_t* xsw     = (bf16_t*)(ws + 0);
  bf16_t* wih_sw  = (bf16_t*)(ws + 16777216);
  bf16_t* whh_sw  = (bf16_t*)(ws + 17039360);
  bf16_t* wout_sw = (bf16_t*)(ws + 17563648);
  float*  xp      = (float*) (ws + 17825792);
  bf16_t* rnn_swz = (bf16_t*)(ws + 84934656);

  pack_x_kernel<<<2048, 256, 0, stream>>>(x, xsw);
  pack_w_kernel<<<32, 256, 0, stream>>>(W_ih, wih_sw, HH, II);
  pack_w_kernel<<<64, 256, 0, stream>>>(W_hh, whh_sw, HH, HH);
  pack_w_kernel<<<32, 256, 0, stream>>>(W_out, wout_sw, OO, HH);

  input_proj_kernel<<<8192, 256, 0, stream>>>(xsw, wih_sw, b_ih, b_hh, xp);

  // single-WG persistent recurrence: h double-buffered in LDS (128 KB)
  rnn_recurrence_kernel<<<1, 1024, 131072, stream>>>(xp, whh_sw, rnn_swz, h_out);

  output_proj_kernel<<<1024, 256, 0, stream>>>(rnn_swz, wout_sw, b_out, y);
}
